// SDS_37099927503047
// MI455X (gfx1250) — compile-verified, matrix operand order checked by reading, not run on hardware
//
#include <hip/hip_runtime.h>
#include <math.h>

#define BB 16
#define CC 96
#define HH 128
#define WW 128
#define HWV 16384           // HH*WW
#define MM 12
#define PAD 3

typedef __attribute__((ext_vector_type(16))) _Float16 v16h;
typedef __attribute__((ext_vector_type(8)))  _Float16 v8h;
typedef __attribute__((ext_vector_type(8)))  float    v8f;

// CDNA5 async DMA: global -> LDS, per-lane addresses, tracked by ASYNCcnt.
__device__ __forceinline__ void async_g2l_b32(unsigned lds_off, const float* g) {
  asm volatile("global_load_async_to_lds_b32 %0, %1, off"
               :: "v"(lds_off), "v"(g) : "memory");
}
__device__ __forceinline__ void wait_asynccnt0() {
  asm volatile("s_wait_asynccnt 0" ::: "memory");
}

// ---------------- K1: depthwise 7x7 conv + bias -> f ----------------
__global__ void dwconv_kernel(const float* __restrict__ x,
                              const float* __restrict__ dww,
                              const float* __restrict__ dwb,
                              float* __restrict__ f) {
  __shared__ float tile[22][22];
  __shared__ float wsm[49];
  const int tx = threadIdx.x & 15, ty = threadIdx.x >> 4;
  const int bx = blockIdx.x & 7, by = blockIdx.x >> 3;   // 8x8 tiles of 16x16
  const int c = blockIdx.y, b = blockIdx.z;
  const float* xp = x + ((size_t)(b * CC + c)) * HWV;
  if (threadIdx.x < 49) wsm[threadIdx.x] = dww[c * 49 + threadIdx.x];
  const bool interior = (bx >= 1) && (bx <= 6) && (by >= 1) && (by <= 6);
  if (interior) {
    // all halo taps in-bounds: async DMA straight into LDS (ASYNCcnt path)
    for (int i = threadIdx.x; i < 22 * 22; i += 256) {
      int ly = i / 22, lx = i % 22;
      int gy = by * 16 + ly - PAD, gx = bx * 16 + lx - PAD;
      async_g2l_b32((unsigned)(size_t)&tile[ly][lx], xp + gy * WW + gx);
    }
    wait_asynccnt0();
  } else {
    for (int i = threadIdx.x; i < 22 * 22; i += 256) {
      int ly = i / 22, lx = i % 22;
      int gy = by * 16 + ly - PAD, gx = bx * 16 + lx - PAD;
      float v = 0.0f;
      if (gy >= 0 && gy < HH && gx >= 0 && gx < WW) v = xp[gy * WW + gx];
      tile[ly][lx] = v;
    }
  }
  __syncthreads();
  float acc = dwb[c];
#pragma unroll
  for (int ky = 0; ky < 7; ky++)
#pragma unroll
    for (int kx = 0; kx < 7; kx++)
      acc = fmaf(tile[ty + ky][tx + kx], wsm[ky * 7 + kx], acc);
  const int gy = by * 16 + ty, gx = bx * 16 + tx;
  f[((size_t)(b * CC + c)) * HWV + gy * WW + gx] = acc;
}

// ---------------- K2: deterministic mean pool over HW ----------------
__global__ void pool_kernel(const float* __restrict__ f, float* __restrict__ pooled) {
  const int bc = blockIdx.x;
  const float* p = f + (size_t)bc * HWV;
  float s = 0.0f;
  for (int i = threadIdx.x; i < HWV; i += 256) s += p[i];
  __shared__ float red[256];
  red[threadIdx.x] = s;
  __syncthreads();
  for (int st = 128; st > 0; st >>= 1) {
    if (threadIdx.x < st) red[threadIdx.x] += red[threadIdx.x + st];
    __syncthreads();
  }
  if (threadIdx.x == 0) pooled[bc] = red[0] * (1.0f / (float)HWV);
}

// ---------------- K3: SE attention MLP ----------------
__global__ void attn_kernel(const float* __restrict__ pooled,
                            const float* __restrict__ w1,
                            const float* __restrict__ w2,
                            float* __restrict__ attn) {
  const int b = blockIdx.x, t = threadIdx.x;
  __shared__ float ps[96], hs[6];
  if (t < 96) ps[t] = pooled[b * 96 + t];
  __syncthreads();
  if (t < 6) {
    float s = 0.0f;
    for (int c = 0; c < 96; c++) s += ps[c] * w1[t * 96 + c];
    hs[t] = fmaxf(s, 0.0f);
  }
  __syncthreads();
  if (t < 96) {
    float s = 0.0f;
#pragma unroll
    for (int r = 0; r < 6; r++) s += hs[r] * w2[t * 6 + r];
    attn[b * 96 + t] = 1.0f / (1.0f + expf(-s));
  }
}

// ---------------- K4: normalize centers -> f16 padded [16][96] ----------------
__global__ void centnorm_kernel(const float* __restrict__ centers, _Float16* __restrict__ cent16) {
  __shared__ float inv[12];
  const int t = threadIdx.x;
  if (t < 12) {
    float s = 0.0f;
    for (int c = 0; c < 96; c++) { float v = centers[t * 96 + c]; s += v * v; }
    inv[t] = 1.0f / fmaxf(sqrtf(s), 1e-12f);
  }
  __syncthreads();
  for (int i = t; i < 16 * 96; i += 128) {
    int m = i / 96, c = i % 96;
    float v = (m < 12) ? centers[m * 96 + c] * inv[m] : 0.0f;
    cent16[i] = (_Float16)v;
  }
}

// ---------------- K5: fused attn*gelu -> LN(C) -> offsets -> bilinear sample, in-place f -> x_bar
__global__ void fuse_kernel(const float* __restrict__ x,
                            const float* __restrict__ Rmap,
                            const float* __restrict__ attn,
                            const float* __restrict__ lng,
                            const float* __restrict__ lnb,
                            const float* __restrict__ offw,
                            float* __restrict__ f) {
  __shared__ float ft[96][65];   // padded stride to dodge bank conflicts
  const int b = blockIdx.y;
  const int n0 = blockIdx.x * 64;
  const int t = threadIdx.x;     // 64 threads; thread t owns pixel column t
  float* fb = f + (size_t)b * CC * HWV;
  for (int i = t; i < 96 * 64; i += 64) {
    int c = i >> 6, p = i & 63;
    float v = fb[(size_t)c * HWV + n0 + p] * attn[b * 96 + c];
    v = 0.5f * v * (1.0f + erff(v * 0.70710678118f));   // exact GELU
    ft[c][p] = v;
  }
  __syncthreads();
  const int p = t;
  const int n = n0 + p;
  const int h = n >> 7, wcol = n & 127;
  float mu = 0.0f;
  for (int c = 0; c < 96; c++) mu += ft[c][p];
  mu *= (1.0f / 96.0f);
  float var = 0.0f;
  for (int c = 0; c < 96; c++) { float d = ft[c][p] - mu; var += d * d; }
  var *= (1.0f / 96.0f);
  const float inv = rsqrtf(var + 1e-5f);
  float ox = 0.0f, oy = 0.0f;
  for (int c = 0; c < 96; c++) {
    float g = (ft[c][p] - mu) * inv * lng[c] + lnb[c];
    ox += offw[c] * g;
    oy += offw[96 + c] * g;
  }
  const float refx = -1.0f + 2.0f * (float)wcol / (float)(WW - 1);
  const float refy = -1.0f + 2.0f * (float)h    / (float)(HH - 1);
  float gx = fminf(fmaxf((refx + ox + 1.0f) * (WW * 0.5f) - 0.5f, 0.0f), (float)(WW - 1));
  float gy = fminf(fmaxf((refy + oy + 1.0f) * (HH * 0.5f) - 0.5f, 0.0f), (float)(HH - 1));
  const float x0f = floorf(gx), y0f = floorf(gy);
  const float wx = gx - x0f, wy = gy - y0f;
  const int x0 = (int)x0f, y0 = (int)y0f;
  const int x1 = min(x0 + 1, WW - 1), y1 = min(y0 + 1, HH - 1);
  const float r00 = Rmap[y0 * WW + x0], r01 = Rmap[y0 * WW + x1];
  const float r10 = Rmap[y1 * WW + x0], r11 = Rmap[y1 * WW + x1];
  const float rb = (r00 * (1.0f - wx) + r01 * wx) * (1.0f - wy) +
                   (r10 * (1.0f - wx) + r11 * wx) * wy;
  const float* xb = x + (size_t)b * CC * HWV;
  for (int c = 0; c < 96; c++) {
    const float* xc = xb + (size_t)c * HWV;
    float v00 = xc[y0 * WW + x0], v01 = xc[y0 * WW + x1];
    float v10 = xc[y1 * WW + x0], v11 = xc[y1 * WW + x1];
    float s = (v00 * (1.0f - wx) + v01 * wx) * (1.0f - wy) +
              (v10 * (1.0f - wx) + v11 * wx) * wy;
    ft[c][p] = s + rb;                 // overwrite own column only
  }
  __syncthreads();
  for (int i = t; i < 96 * 64; i += 64) {
    int c = i >> 6, pp = i & 63;
    fb[(size_t)c * HWV + n0 + pp] = ft[c][pp];   // coalesced store, in-place x_bar
  }
}

// ---------------- K6: WMMA center similarity + argmax -> group_idx ----------------
// Pixel-major f16 staging: each lane's A-fragment is two contiguous 16B runs
// -> 2x ds_load_b128 per fragment (instead of 16x ds_load_u16).
// Row stride 104 halves = 208B: 16B-aligned, gcd(52,64)=4 banking.
__global__ void sim_kernel(const float* __restrict__ x,
                           const _Float16* __restrict__ cent,
                           int* __restrict__ gid) {
  __shared__ _Float16 xt2[128][104];
  __shared__ float simt[8][16][16];
  const int b = blockIdx.y;
  const int n0 = blockIdx.x * 128;
  const int t = threadIdx.x;
  const float* xb = x + (size_t)b * CC * HWV;
  for (int i = t; i < 96 * 128; i += 256) {
    int c = i >> 7, p = i & 127;
    xt2[p][c] = (_Float16)xb[(size_t)c * HWV + n0 + p];
  }
  __syncthreads();
  const int wv = t >> 5, lane = t & 31;
  const int col = lane & 15;                 // M-row for A, N-col for B
  const int kb8 = (lane >= 16) ? 1 : 0;      // (kbase=8 for upper half-wave) / 8
  const int pix = (wv << 4) + col;
  const v8h* arow = (const v8h*)&xt2[pix][0];
  const v8h* brow = (const v8h*)&cent[col * 96];
  v8f acc = {};
#pragma unroll
  for (int s = 0; s < 3; s++) {
    union { v16h v; v8h h[2]; } ua, ub;
    ua.h[0] = arow[4 * s + kb8];       // K = 32s + kbase + 0..7
    ua.h[1] = arow[4 * s + kb8 + 2];   // K = 32s + kbase + 16..23
    ub.h[0] = brow[4 * s + kb8];
    ub.h[1] = brow[4 * s + kb8 + 2];
    acc = __builtin_amdgcn_wmma_f32_16x16x32_f16(false, ua.v, false, ub.v,
                                                 (short)0, acc, false, false);
  }
  const int rbase = (lane < 16) ? 0 : 8;
#pragma unroll
  for (int v = 0; v < 8; v++) simt[wv][rbase + v][col] = acc[v];
  __syncthreads();
  if (lane < 16) {
    float best = simt[wv][lane][0];
    int bi = 0;
#pragma unroll
    for (int m = 1; m < 12; m++) {
      float vv = simt[wv][lane][m];
      if (vv > best) { best = vv; bi = m; }   // strict >: first-max like jnp.argmax
    }
    gid[b * HWV + n0 + (wv << 4) + lane] = bi;
  }
}

// ---------------- K7: stable counting sort (12 bins) per batch -> perm ----------------
__global__ void sort_kernel(const int* __restrict__ gid, int* __restrict__ perm) {
  const int b = blockIdx.x;
  const int* g = gid + b * HWV;
  int* pm = perm + b * HWV;
  __shared__ int cnt[16];
  __shared__ int offs[16];
  __shared__ int wcnt[8];
  const int t = threadIdx.x;
  if (t < 16) cnt[t] = 0;
  __syncthreads();
  for (int n = t; n < HWV; n += 256) atomicAdd(&cnt[g[n]], 1);
  __syncthreads();
  if (t == 0) {
    int run = 0;
    for (int m = 0; m < 12; m++) { offs[m] = run; run += cnt[m]; }
  }
  __syncthreads();
  const int wv = t >> 5, lane = t & 31;
  for (int m = 0; m < 12; m++) {
    int running = offs[m];
    for (int c0 = 0; c0 < HWV; c0 += 256) {
      const int n = c0 + t;
      const bool match = (g[n] == m);
      const unsigned mask32 = (unsigned)__ballot(match);     // wave32: low 32 bits
      const int lpref = __popc(mask32 & ((1u << lane) - 1u));
      if (lane == 0) wcnt[wv] = __popc(mask32);
      __syncthreads();
      int wpref = 0, tot = 0;
#pragma unroll
      for (int q = 0; q < 8; q++) { int c = wcnt[q]; if (q < wv) wpref += c; tot += c; }
      if (match) pm[running + wpref + lpref] = n;
      running += tot;
      __syncthreads();
    }
  }
}

// ---------------- K8: permuted gather -> output_map ----------------
__global__ void gather_kernel(const float* __restrict__ xbar,
                              const int* __restrict__ perm,
                              float* __restrict__ out) {
  const size_t e = (size_t)blockIdx.x * 256 + threadIdx.x;
  const int i = (int)(e & (HWV - 1));
  const size_t bc = e >> 14;               // /HWV
  const int b = (int)(bc / CC);
  __builtin_prefetch(&perm[b * HWV + ((i + 256) & (HWV - 1))], 0, 0);
  const int src = perm[b * HWV + i];
  out[e] = xbar[bc * HWV + src];
}

// ---------------- K9: tail outputs: perm indices (as float) + centers ----------------
__global__ void tail_kernel(const int* __restrict__ perm,
                            const float* __restrict__ centers,
                            float* __restrict__ out_tail) {
  const int i = blockIdx.x * 256 + threadIdx.x;
  const int NP = BB * HWV;
  if (i < NP) out_tail[i] = (float)perm[i];
  else if (i < NP + MM * CC) out_tail[i] = centers[i - NP];
}

extern "C" void kernel_launch(void* const* d_in, const int* in_sizes, int n_in,
                              void* d_out, int out_size, void* d_ws, size_t ws_size,
                              hipStream_t stream) {
  const float* x       = (const float*)d_in[0];
  const float* dw_w    = (const float*)d_in[1];
  const float* dw_b    = (const float*)d_in[2];
  const float* ca_w1   = (const float*)d_in[3];
  const float* ca_w2   = (const float*)d_in[4];
  const float* ln_g    = (const float*)d_in[5];
  const float* ln_b    = (const float*)d_in[6];
  const float* off_w   = (const float*)d_in[7];
  const float* Rmap    = (const float*)d_in[8];
  const float* centers = (const float*)d_in[9];
  float* out = (float*)d_out;

  // workspace layout (256B-aligned regions)
  char* ws = (char*)d_ws;
  const size_t F_BYTES = (size_t)BB * CC * HWV * sizeof(float);    // 100663296
  float*     f      = (float*)(ws);                                // f, later x_bar (in-place)
  float*     pooled = (float*)(ws + F_BYTES);                      // BB*CC
  float*     attn   = (float*)(ws + F_BYTES + 6144);               // BB*CC
  _Float16*  cent16 = (_Float16*)(ws + F_BYTES + 12288);           // 16*96 f16
  int*       gid    = (int*)(ws + F_BYTES + 16384);                // BB*HWV
  int*       perm   = (int*)(ws + F_BYTES + 16384 + (size_t)BB * HWV * 4);

  dwconv_kernel<<<dim3(64, CC, BB), 256, 0, stream>>>(x, dw_w, dw_b, f);
  pool_kernel<<<BB * CC, 256, 0, stream>>>(f, pooled);
  attn_kernel<<<BB, 128, 0, stream>>>(pooled, ca_w1, ca_w2, attn);
  centnorm_kernel<<<1, 128, 0, stream>>>(centers, cent16);
  fuse_kernel<<<dim3(HWV / 64, BB), 64, 0, stream>>>(x, Rmap, attn, ln_g, ln_b, off_w, f);
  sim_kernel<<<dim3(HWV / 128, BB), 256, 0, stream>>>(x, cent16, gid);
  sort_kernel<<<BB, 256, 0, stream>>>(gid, perm);
  gather_kernel<<<(BB * CC * HWV) / 256, 256, 0, stream>>>(f, perm, out);
  const size_t OMAP = (size_t)BB * CC * HWV;
  tail_kernel<<<(BB * HWV + MM * CC + 255) / 256, 256, 0, stream>>>(perm, centers, out + OMAP);
}
